// MultiHeadSelfAttention_13855564497606
// MI455X (gfx1250) — matrix operand order repaired and checked by reading, not compiled, not run on hardware
//
#include <hip/hip_runtime.h>
#include <hip/hip_bf16.h>

// MI455X (gfx1250) multi-head self-attention forward.
// bf16 WMMA (V_WMMA_F32_16X16X32_BF16) for all four matmuls.
// x / qkv_w / out_w are pre-converted to bf16 ONCE; GEMM waves compute
// 32(M) x 64(N) tiles so each 4-B-fragment load feeds 8 WMMAs.

typedef __attribute__((ext_vector_type(16))) __bf16 v16bf;
typedef __attribute__((ext_vector_type(8)))  __bf16 v8bf;
typedef __attribute__((ext_vector_type(8)))  float  v8f;
typedef __attribute__((ext_vector_type(4)))  float  v4f;

#define B_   4
#define S_   2048
#define HID  1024
#define NH   16
#define DH   64

__device__ __forceinline__ v8f zero8() {
  v8f z;
#pragma unroll
  for (int i = 0; i < 8; ++i) z[i] = 0.0f;
  return z;
}

__device__ __forceinline__ v8f wmma_bf16(v16bf a, v16bf b, v8f c) {
  // (neg_a, A, neg_b, B, c_mod, C, reuse_a, reuse_b)
  return __builtin_amdgcn_wmma_f32_16x16x32_bf16(false, a, false, b, (short)0, c,
                                                 false, false);
}

__device__ __forceinline__ v16bf load_afrag(const __bf16* row, int k0, int kb8) {
  // A fragment: K = k0 + 2*kb8 + {0..15}
  v8bf lo = *(const v8bf*)(row + k0 + 2 * kb8);
  v8bf hi = *(const v8bf*)(row + k0 + 2 * kb8 + 8);
  v16bf a;
#pragma unroll
  for (int e = 0; e < 8; ++e) { a[e] = lo[e]; a[e + 8] = hi[e]; }
  return a;
}

// ---------------------------------------------------------------------------
// Kernel 0: one-time fp32 -> bf16 conversion (8 elements/thread, vectorized)
// ---------------------------------------------------------------------------
__global__ void __launch_bounds__(256)
cvt_f32_bf16(const float* __restrict__ src, __bf16* __restrict__ dst, int n) {
  const int i = (blockIdx.x * 256 + threadIdx.x) * 8;
  if (i < n) {
    v4f a = *(const v4f*)(src + i);
    v4f b = *(const v4f*)(src + i + 4);
    v8bf o;
#pragma unroll
    for (int e = 0; e < 4; ++e) {
      o[e]     = (__bf16)a[e];
      o[e + 4] = (__bf16)b[e];
    }
    *(v8bf*)(dst + i) = o;
  }
}

// ---------------------------------------------------------------------------
// Kernel 1: qkv = x @ qkv_w^T + qkv_b   (all-bf16 WMMA, f32 accumulate)
// Each wave: 32(M) x 64(N) tile; 8 WMMAs per 32-wide K step reusing the
// 4 B-fragments across 2 A-fragments.
// q scaled by 1/sqrt(64)=0.125 at store; v stored transposed [B,h,d,S].
// ---------------------------------------------------------------------------
__global__ void __launch_bounds__(32)
qkv_gemm(const __bf16* __restrict__ xb, const __bf16* __restrict__ wb,
         const float* __restrict__ bias,
         __bf16* __restrict__ qb, __bf16* __restrict__ kbuf,
         __bf16* __restrict__ vt) {
  const int lane = threadIdx.x;
  const int col  = lane & 15;        // N within 16-tile / A-row m
  const int half = lane >> 4;        // lane half
  const int kb8  = half * 8;         // A-frag K sub-base
  const int m0   = blockIdx.y * 32;  // row in flattened B*S
  const int n0   = blockIdx.x * 64;  // col in 3*HID

  const __bf16* arow0 = xb + (size_t)(m0 + col) * HID;
  const __bf16* arow1 = xb + (size_t)(m0 + 16 + col) * HID;

  v8f acc[2][4];
#pragma unroll
  for (int u = 0; u < 2; ++u)
#pragma unroll
    for (int t = 0; t < 4; ++t) acc[u][t] = zero8();

#pragma unroll 2
  for (int k0 = 0; k0 < HID; k0 += 32) {
    v16bf a0 = load_afrag(arow0, k0, kb8);
    v16bf a1 = load_afrag(arow1, k0, kb8);

    const int kbB = k0 + half * 16;
    v16bf bfrag[4];
#pragma unroll
    for (int t = 0; t < 4; ++t)
      bfrag[t] = *(const v16bf*)(wb + (size_t)(n0 + t * 16 + col) * HID + kbB);
#pragma unroll
    for (int t = 0; t < 4; ++t) {
      acc[0][t] = wmma_bf16(a0, bfrag[t], acc[0][t]);
      acc[1][t] = wmma_bf16(a1, bfrag[t], acc[1][t]);
    }
  }

  const int mbase = half * 8;  // C/D: M = vgpr + 8*half, N = col
#pragma unroll
  for (int u = 0; u < 2; ++u) {
#pragma unroll
    for (int t = 0; t < 4; ++t) {
      const int gn = n0 + t * 16 + col;
      const float bvv = bias[gn];
#pragma unroll
      for (int r = 0; r < 8; ++r) {
        const int gm = m0 + u * 16 + mbase + r;
        const int b  = gm / S_;
        const int s  = gm - b * S_;
        const float val = acc[u][t][r] + bvv;
        if (gn < HID) {
          const int h = gn / DH, d = gn % DH;
          qb[(((size_t)b * NH + h) * S_ + s) * DH + d] = (__bf16)(val * 0.125f);
        } else if (gn < 2 * HID) {
          const int c = gn - HID, h = c / DH, d = c % DH;
          kbuf[(((size_t)b * NH + h) * S_ + s) * DH + d] = (__bf16)val;
        } else {
          const int c = gn - 2 * HID, h = c / DH, d = c % DH;
          vt[(((size_t)b * NH + h) * DH + d) * S_ + s] = (__bf16)val;  // transposed
        }
      }
    }
  }
}

// ---------------------------------------------------------------------------
// Kernel 2: flash attention. One wave per (b,h, 16-row q tile).
// kv processed 32 at a time: scores via 4 WMMAs, online softmax, P staged
// through LDS (C/D layout -> A layout), ctx += P*V via 4 WMMAs.
// ---------------------------------------------------------------------------
__global__ void __launch_bounds__(32)
attention(const __bf16* __restrict__ qb, const __bf16* __restrict__ kbuf,
          const __bf16* __restrict__ vt, const float* __restrict__ amask,
          __bf16* __restrict__ ctx) {
  __shared__ __align__(16) __bf16 ldsP[16 * 32];

  const int lane  = threadIdx.x;
  const int col   = lane & 15;
  const int half  = lane >> 4;
  const int kb8   = half * 8;
  const int mbase = half * 8;

  const int bh = blockIdx.x;  // b*NH + h
  const int b  = bh >> 4;
  const int h  = bh & 15;
  const int q0 = blockIdx.y * 16;

  // Q A-fragments for the two 32-wide d-steps (held in registers all loop)
  const __bf16* Qrow = qb + ((size_t)bh * S_ + q0 + col) * DH;
  v16bf qa[2];
#pragma unroll
  for (int ks = 0; ks < 2; ++ks) qa[ks] = load_afrag(Qrow, ks * 32, kb8);

  float mrow[8], lrow[8];
#pragma unroll
  for (int r = 0; r < 8; ++r) { mrow[r] = -3.0e38f; lrow[r] = 0.0f; }
  v8f o[4];
#pragma unroll
  for (int t = 0; t < 4; ++t) o[t] = zero8();

  const __bf16* Kbase = kbuf + (size_t)bh * S_ * DH;
  const __bf16* Vbase = vt + (size_t)bh * DH * S_;
  const float*  mptr  = amask + (size_t)b * S_;

  for (int kv0 = 0; kv0 < S_; kv0 += 32) {
    if (kv0 + 32 < S_) {  // global_prefetch_b8 of next K/V tiles
      __builtin_prefetch(Kbase + (size_t)(kv0 + 32 + col) * DH, 0, 3);
      __builtin_prefetch(Vbase + (size_t)col * S_ + kv0 + 32, 0, 3);
    }

    // scores: two 16x16 tiles over kv0..kv0+31
    v8f s0 = zero8(), s1 = zero8();
#pragma unroll
    for (int ks = 0; ks < 2; ++ks) {
      v16bf k0 = *(const v16bf*)(Kbase + (size_t)(kv0 + col) * DH + ks * 32 + half * 16);
      s0 = wmma_bf16(qa[ks], k0, s0);
      v16bf k1 = *(const v16bf*)(Kbase + (size_t)(kv0 + 16 + col) * DH + ks * 32 + half * 16);
      s1 = wmma_bf16(qa[ks], k1, s1);
    }
    const float bias0 = (1.0f - mptr[kv0 + col]) * -1.0e9f;
    const float bias1 = (1.0f - mptr[kv0 + 16 + col]) * -1.0e9f;

    // online softmax; row stats reduced across the 16-lane half that owns M
#pragma unroll
    for (int r = 0; r < 8; ++r) {
      float v0 = s0[r] + bias0;
      float v1 = s1[r] + bias1;
      float mx = fmaxf(v0, v1);
#pragma unroll
      for (int msk = 1; msk <= 8; msk <<= 1) mx = fmaxf(mx, __shfl_xor(mx, msk, 32));
      const float mnew = fmaxf(mrow[r], mx);
      const float corr = __expf(mrow[r] - mnew);
      const float p0 = __expf(v0 - mnew);
      const float p1 = __expf(v1 - mnew);
      float rs = p0 + p1;
#pragma unroll
      for (int msk = 1; msk <= 8; msk <<= 1) rs += __shfl_xor(rs, msk, 32);
      lrow[r] = lrow[r] * corr + rs;
      mrow[r] = mnew;
#pragma unroll
      for (int t = 0; t < 4; ++t) o[t][r] *= corr;
      ldsP[(mbase + r) * 32 + col]      = (__bf16)p0;
      ldsP[(mbase + r) * 32 + 16 + col] = (__bf16)p1;
    }
    __syncthreads();  // single-wave WG: cheap; orders DS write->read

    v16bf pa;
    {
      v8bf lo = *(const v8bf*)(&ldsP[col * 32 + 2 * kb8]);
      v8bf hi = *(const v8bf*)(&ldsP[col * 32 + 2 * kb8 + 8]);
#pragma unroll
      for (int e = 0; e < 8; ++e) { pa[e] = lo[e]; pa[e + 8] = hi[e]; }
    }
    __syncthreads();

    // ctx += P(16x32) * V(32x64); V transposed so each lane load is contiguous
#pragma unroll
    for (int t = 0; t < 4; ++t) {
      v16bf vb = *(const v16bf*)(Vbase + (size_t)(t * 16 + col) * S_ + kv0 + half * 16);
      o[t] = wmma_bf16(pa, vb, o[t]);
    }
  }

  // normalize, store ctx as [B, S, h*64+d] so the out-proj reads it row-major
#pragma unroll
  for (int t = 0; t < 4; ++t) {
#pragma unroll
    for (int r = 0; r < 8; ++r) {
      const float val = o[t][r] / lrow[r];
      ctx[((size_t)b * S_ + q0 + mbase + r) * HID + h * DH + t * 16 + col] =
          (__bf16)val;
    }
  }
}

// ---------------------------------------------------------------------------
// Kernel 3: out = ctx @ out_w^T + out_b   (all-bf16 WMMA, f32 out)
// Same 32x64-per-wave tiling as qkv_gemm.
// ---------------------------------------------------------------------------
__global__ void __launch_bounds__(32)
out_gemm(const __bf16* __restrict__ ctx, const __bf16* __restrict__ wb,
         const float* __restrict__ bias, float* __restrict__ out) {
  const int lane = threadIdx.x;
  const int col  = lane & 15;
  const int half = lane >> 4;
  const int kb8  = half * 8;
  const int m0   = blockIdx.y * 32;
  const int n0   = blockIdx.x * 64;

  const __bf16* arow0 = ctx + (size_t)(m0 + col) * HID;
  const __bf16* arow1 = ctx + (size_t)(m0 + 16 + col) * HID;

  v8f acc[2][4];
#pragma unroll
  for (int u = 0; u < 2; ++u)
#pragma unroll
    for (int t = 0; t < 4; ++t) acc[u][t] = zero8();

#pragma unroll 2
  for (int k0 = 0; k0 < HID; k0 += 32) {
    v16bf a0 = load_afrag(arow0, k0, kb8);
    v16bf a1 = load_afrag(arow1, k0, kb8);

    const int kbB = k0 + half * 16;
    v16bf bfrag[4];
#pragma unroll
    for (int t = 0; t < 4; ++t)
      bfrag[t] = *(const v16bf*)(wb + (size_t)(n0 + t * 16 + col) * HID + kbB);
#pragma unroll
    for (int t = 0; t < 4; ++t) {
      acc[0][t] = wmma_bf16(a0, bfrag[t], acc[0][t]);
      acc[1][t] = wmma_bf16(a1, bfrag[t], acc[1][t]);
    }
  }

  const int mbase = half * 8;
#pragma unroll
  for (int u = 0; u < 2; ++u) {
#pragma unroll
    for (int t = 0; t < 4; ++t) {
      const int gn = n0 + t * 16 + col;
      const float bvv = bias[gn];
#pragma unroll
      for (int r = 0; r < 8; ++r) {
        out[(size_t)(m0 + u * 16 + mbase + r) * HID + gn] = acc[u][t][r] + bvv;
      }
    }
  }
}

// ---------------------------------------------------------------------------
extern "C" void kernel_launch(void* const* d_in, const int* in_sizes, int n_in,
                              void* d_out, int out_size, void* d_ws, size_t ws_size,
                              hipStream_t stream) {
  const float* x    = (const float*)d_in[0];
  const float* mask = (const float*)d_in[1];
  const float* qkvw = (const float*)d_in[2];
  const float* qkvb = (const float*)d_in[3];
  const float* outw = (const float*)d_in[4];
  const float* outb = (const float*)d_in[5];
  float* out = (float*)d_out;

  // Workspace layout (bytes):
  //   [0, 16M)   q   bf16 [B,h,S,d]      [16M,32M)  k   bf16 [B,h,S,d]
  //   [32M,48M)  v^T bf16 [B,h,d,S]      [48M,64M)  ctx bf16 [B,S,H]
  //   [64M,80M)  xb  bf16 [B*S,H]
  //   [80M,86M)  qkv_w bf16 [3H,H]       [86M,88M)  out_w bf16 [H,H]
  char* ws = (char*)d_ws;
  const size_t seg = (size_t)B_ * NH * S_ * DH * sizeof(__bf16);  // 16 MiB
  __bf16* qb    = (__bf16*)(ws);
  __bf16* kbuf  = (__bf16*)(ws + seg);
  __bf16* vt    = (__bf16*)(ws + 2 * seg);
  __bf16* ctx   = (__bf16*)(ws + 3 * seg);
  __bf16* xb    = (__bf16*)(ws + 4 * seg);
  __bf16* wqkvb = (__bf16*)(ws + 5 * seg);
  __bf16* woutb = (__bf16*)(ws + 5 * seg + (size_t)3 * HID * HID * sizeof(__bf16));

  const int n_x    = B_ * S_ * HID;     // 8,388,608
  const int n_wqkv = 3 * HID * HID;     // 3,145,728
  const int n_wout = HID * HID;         // 1,048,576

  dim3 cblk(256);
  cvt_f32_bf16<<<dim3(n_x / (8 * 256)), cblk, 0, stream>>>(x, xb, n_x);
  cvt_f32_bf16<<<dim3(n_wqkv / (8 * 256)), cblk, 0, stream>>>(qkvw, wqkvb, n_wqkv);
  cvt_f32_bf16<<<dim3(n_wout / (8 * 256)), cblk, 0, stream>>>(outw, woutb, n_wout);

  dim3 blk(32);
  qkv_gemm<<<dim3(3 * HID / 64, (B_ * S_) / 32), blk, 0, stream>>>(
      xb, wqkvb, qkvb, qb, kbuf, vt);
  attention<<<dim3(B_ * NH, S_ / 16), blk, 0, stream>>>(qb, kbuf, vt, mask, ctx);
  out_gemm<<<dim3(HID / 64, (B_ * S_) / 32), blk, 0, stream>>>(ctx, woutb, outb, out);
}
